// VGG3_Test_B1_60430189855192
// MI455X (gfx1250) — 8-bit matrix operand order repaired and checked by reading, compiled, not run on hardware
//
#include <hip/hip_runtime.h>

typedef int v8i __attribute__((ext_vector_type(8)));

#define BATCH 4096

__global__ void prep_conv2w(const float* __restrict__ w, signed char* __restrict__ dst) {
  int i = blockIdx.x * 256 + threadIdx.x;
  if (i >= 9 * 64 * 64) return;
  int ci = i & 63;
  int co = (i >> 6) & 63;
  int tap = i >> 12;
  float v = w[(co * 64 + ci) * 9 + tap];
  dst[(tap * 64 + co) * 64 + ci] = (v >= 0.f) ? 1 : -1;
}

__global__ void binarize_w(const float* __restrict__ w, signed char* __restrict__ dst, int n) {
  int i = blockIdx.x * 256 + threadIdx.x;
  if (i < n) dst[i] = (w[i] >= 0.f) ? 1 : -1;
}

__global__ __launch_bounds__(256) void conv1_pool_bn_bin(
    const float* __restrict__ x, const float* __restrict__ w, const float* __restrict__ bias,
    const float* __restrict__ g, const float* __restrict__ be,
    const float* __restrict__ mn, const float* __restrict__ vr,
    signed char* __restrict__ act1) {
  int idx = blockIdx.x * 256 + threadIdx.x;
  int c = idx & 63;
  int t = idx >> 6;
  int px = t % 14; t /= 14;
  int py = t % 14; t /= 14;
  int b = t;
  if (b >= BATCH) return;
  float ws[9];
#pragma unroll
  for (int k = 0; k < 9; k++) ws[k] = (w[c * 9 + k] >= 0.f) ? 1.f : -1.f;
  const float* xb = x + (size_t)b * 28 * 28;
  float m = -3.4e38f;
#pragma unroll
  for (int i = 0; i < 2; i++)
#pragma unroll
    for (int j = 0; j < 2; j++) {
      int cy = 2 * py + i, cx = 2 * px + j;
      float s = 0.f;
#pragma unroll
      for (int dy = 0; dy < 3; dy++) {
        int Y = cy - 1 + dy;
#pragma unroll
        for (int dx = 0; dx < 3; dx++) {
          int X = cx - 1 + dx;
          float xv = (Y >= 0 && Y < 28 && X >= 0 && X < 28) ? xb[Y * 28 + X] : 0.f;
          s = fmaf(xv, ws[dy * 3 + dx], s);
        }
      }
      m = fmaxf(m, s);
    }
  m += bias[c];
  float inv = g[c] * rsqrtf(vr[c] + 1e-5f);
  float y = m * inv + (be[c] - mn[c] * inv);
  act1[(((size_t)b * 16 + (py + 1)) * 16 + (px + 1)) * 64 + c] = (y >= 0.f) ? 1 : -1;
}

__global__ __launch_bounds__(128) void conv2_wmma(
    const signed char* __restrict__ act1, const signed char* __restrict__ w8,
    const float* __restrict__ bias, const float* __restrict__ g, const float* __restrict__ be,
    const float* __restrict__ mn, const float* __restrict__ vr,
    signed char* __restrict__ act2) {
  __shared__ float lds[196 * 64];
  int b = blockIdx.x;
  int lane = threadIdx.x & 31;
  int wv = threadIdx.x >> 5;
  int l15 = lane & 15;
  int hi = (lane >= 16) ? 1 : 0;

  v8i bf[9];
  {
    const signed char* wp = w8 + (wv * 16 + l15) * 64 + hi * 16;
#pragma unroll
    for (int tap = 0; tap < 9; tap++) {
      int4 x0 = *(const int4*)(wp + tap * 4096);
      int4 x1 = *(const int4*)(wp + tap * 4096 + 32);
      bf[tap][0] = x0.x; bf[tap][1] = x0.y; bf[tap][2] = x0.z; bf[tap][3] = x0.w;
      bf[tap][4] = x1.x; bf[tap][5] = x1.y; bf[tap][6] = x1.z; bf[tap][7] = x1.w;
    }
  }

  const signed char* ab = act1 + (size_t)b * (16 * 16 * 64) + hi * 16;
  for (int t = 0; t < 13; t++) {
    int m = t * 16 + l15;
    int p = (m < 196) ? m : 195;
    int cy = p / 14, cx = p - cy * 14;
    v8i acc = {};
#pragma unroll
    for (int tap = 0; tap < 9; tap++) {
      int dy = tap / 3, dx = tap - dy * 3;
      const signed char* pa = ab + ((cy + dy) * 16 + (cx + dx)) * 64;
      v8i a;
      const int* q;
      q = (const int*)(pa);      a[0] = q[0]; a[1] = q[1]; a[2] = q[2]; a[3] = q[3];
      q = (const int*)(pa + 32); a[4] = q[0]; a[5] = q[1]; a[6] = q[2]; a[7] = q[3];
      acc = __builtin_amdgcn_wmma_i32_16x16x64_iu8(true, a, true, bf[tap], acc, false, false);
    }
#pragma unroll
    for (int r = 0; r < 8; r++) {
      int mg = t * 16 + r + hi * 8;
      if (mg < 196) lds[mg * 64 + wv * 16 + l15] = (float)acc[r];
    }
  }
  __syncthreads();

  for (int i = threadIdx.x; i < 49 * 64; i += 128) {
    int co = i & 63;
    int pp = i >> 6;
    int py = pp / 7, px = pp - py * 7;
    int base = ((2 * py) * 14 + 2 * px) * 64 + co;
    float m = fmaxf(fmaxf(lds[base], lds[base + 64]),
                    fmaxf(lds[base + 14 * 64], lds[base + 14 * 64 + 64]));
    m += bias[co];
    float inv = g[co] * rsqrtf(vr[co] + 1e-5f);
    float y = m * inv + (be[co] - mn[co] * inv);
    act2[(size_t)b * 3136 + co * 49 + py * 7 + px] = (y >= 0.f) ? 1 : -1;
  }
}

__global__ __launch_bounds__(128) void fc1_wmma(
    const signed char* __restrict__ act2, const signed char* __restrict__ w8,
    const float* __restrict__ bias, const float* __restrict__ g, const float* __restrict__ be,
    const float* __restrict__ mn, const float* __restrict__ vr,
    signed char* __restrict__ act3) {
  int mg = blockIdx.x;
  int ng = blockIdx.y;
  int lane = threadIdx.x & 31;
  int wv = threadIdx.x >> 5;
  int l15 = lane & 15;
  int hi = (lane >= 16) ? 1 : 0;
  int n = ng * 64 + wv * 16 + l15;
  int row = mg * 16 + l15;
  const signed char* pa = act2 + (size_t)row * 3136 + hi * 16;
  const signed char* pb = w8 + (size_t)n * 3136 + hi * 16;

  v8i acc = {};
  for (int k0 = 0; k0 < 3136; k0 += 64) {
    __builtin_prefetch(pa + k0 + 512, 0, 1);
    __builtin_prefetch(pb + k0 + 512, 0, 1);
    v8i a, bm;
    const int* q;
    q = (const int*)(pa + k0);      a[0] = q[0]; a[1] = q[1]; a[2] = q[2]; a[3] = q[3];
    q = (const int*)(pa + k0 + 32); a[4] = q[0]; a[5] = q[1]; a[6] = q[2]; a[7] = q[3];
    int4 x0 = *(const int4*)(pb + k0);
    int4 x1 = *(const int4*)(pb + k0 + 32);
    bm[0] = x0.x; bm[1] = x0.y; bm[2] = x0.z; bm[3] = x0.w;
    bm[4] = x1.x; bm[5] = x1.y; bm[6] = x1.z; bm[7] = x1.w;
    acc = __builtin_amdgcn_wmma_i32_16x16x64_iu8(true, a, true, bm, acc, false, false);
  }

  float inv = g[n] * rsqrtf(vr[n] + 1e-5f);
  float sh = be[n] - mn[n] * inv;
  float bb = bias[n];
#pragma unroll
  for (int r = 0; r < 8; r++) {
    int rr = mg * 16 + r + hi * 8;
    float y = ((float)acc[r] + bb) * inv + sh;
    act3[(size_t)rr * 2048 + n] = (y >= 0.f) ? 1 : -1;
  }
}

__global__ __launch_bounds__(256) void fc2_kernel(
    const signed char* __restrict__ act3, const signed char* __restrict__ w8,
    const float* __restrict__ bias, const float* __restrict__ scale, float* __restrict__ out) {
  int idx = blockIdx.x * 256 + threadIdx.x;
  if (idx >= BATCH * 10) return;
  int n = idx % 10, b = idx / 10;
  const signed char* a = act3 + (size_t)b * 2048;
  const signed char* w = w8 + (size_t)n * 2048;
  int s = 0;
  for (int k = 0; k < 2048; k += 4) {
    int av = *(const int*)(a + k);
    int wv = *(const int*)(w + k);
#pragma unroll
    for (int j = 0; j < 4; j++)
      s += (int)(signed char)(av >> (8 * j)) * (int)(signed char)(wv >> (8 * j));
  }
  out[idx] = ((float)s + bias[n]) * scale[0];
}

extern "C" void kernel_launch(void* const* d_in, const int* in_sizes, int n_in,
                              void* d_out, int out_size, void* d_ws, size_t ws_size,
                              hipStream_t stream) {
  (void)in_sizes; (void)n_in; (void)out_size; (void)ws_size;
  const float* x    = (const float*)d_in[0];
  const float* c1w  = (const float*)d_in[1];
  const float* c1b  = (const float*)d_in[2];
  const float* c2w  = (const float*)d_in[3];
  const float* c2b  = (const float*)d_in[4];
  const float* f1w  = (const float*)d_in[5];
  const float* f1b  = (const float*)d_in[6];
  const float* f2w  = (const float*)d_in[7];
  const float* f2b  = (const float*)d_in[8];
  const float* bn1g = (const float*)d_in[9],  *bn1b = (const float*)d_in[10];
  const float* bn1m = (const float*)d_in[11], *bn1v = (const float*)d_in[12];
  const float* bn2g = (const float*)d_in[13], *bn2b = (const float*)d_in[14];
  const float* bn2m = (const float*)d_in[15], *bn2v = (const float*)d_in[16];
  const float* bn3g = (const float*)d_in[17], *bn3b = (const float*)d_in[18];
  const float* bn3m = (const float*)d_in[19], *bn3v = (const float*)d_in[20];
  const float* scl  = (const float*)d_in[21];

  char* ws = (char*)d_ws;
  signed char* c2w8 = (signed char*)(ws + 0);
  signed char* f1w8 = (signed char*)(ws + 36864);
  signed char* f2w8 = (signed char*)(ws + 6459392);
  signed char* act1 = (signed char*)(ws + 6479872);
  signed char* act2 = (signed char*)(ws + 73588736);
  signed char* act3 = (signed char*)(ws + 86433792);

  hipMemsetAsync(act1, 0, (size_t)BATCH * 16 * 16 * 64, stream);
  prep_conv2w<<<(9 * 64 * 64 + 255) / 256, 256, 0, stream>>>(c2w, c2w8);
  binarize_w<<<(2048 * 3136 + 255) / 256, 256, 0, stream>>>(f1w, f1w8, 2048 * 3136);
  binarize_w<<<(10 * 2048 + 255) / 256, 256, 0, stream>>>(f2w, f2w8, 10 * 2048);

  conv1_pool_bn_bin<<<(BATCH * 14 * 14 * 64) / 256, 256, 0, stream>>>(
      x, c1w, c1b, bn1g, bn1b, bn1m, bn1v, act1);
  conv2_wmma<<<BATCH, 128, 0, stream>>>(act1, c2w8, c2b, bn2g, bn2b, bn2m, bn2v, act2);
  fc1_wmma<<<dim3(256, 32), 128, 0, stream>>>(act2, f1w8, f1b, bn3g, bn3b, bn3m, bn3v, act3);
  fc2_kernel<<<(BATCH * 10 + 255) / 256, 256, 0, stream>>>(act3, f2w8, f2b, scl, (float*)d_out);
}
